// Eq2NetSet_30090540875886
// MI455X (gfx1250) — compile-verified, matrix operand order checked by reading, not run on hardware
//
#include <hip/hip_runtime.h>
#include <stdint.h>

// ---------------------------------------------------------------------------
// Eq2NetSet fused pair-MLP for MI455X (gfx1250, wave32, WMMA f16 + TDM)
// B=4, N=256, D=64, H=128.  ~21.5 GFLOP on <1MB of real inputs -> fully
// fused; [B,N,N,*] tensors never touch memory.  Per layer: preload the whole
// B panel (<=256 VGPRs) with ds_load_b128, then fire all WMMAs as one
// hazard-free burst.  Weight staging via Tensor Data Mover.
// ---------------------------------------------------------------------------

typedef __attribute__((ext_vector_type(16))) _Float16     v16h;
typedef __attribute__((ext_vector_type(8)))  _Float16     v8h;
typedef __attribute__((ext_vector_type(8)))  float        v8f;
typedef __attribute__((ext_vector_type(4)))  unsigned int v4u;
typedef __attribute__((ext_vector_type(8)))  int          v8i;
typedef __attribute__((ext_vector_type(4)))  int          v4i;

#define LDH 136   // activation leading dim (elems); 272B rows stay 16B-aligned

#if __has_builtin(__builtin_amdgcn_tensor_load_to_lds) && \
    __has_builtin(__builtin_amdgcn_s_wait_tensorcnt)
#define USE_TDM 1
#else
#define USE_TDM 0
#endif

static __device__ __forceinline__ v8f wmma_f16(v16h a, v16h b, v8f c) {
  // D = A(16x32 f16) * B(32x16 f16) + C(16x16 f32)
  return __builtin_amdgcn_wmma_f32_16x16x32_f16(
      false, a, false, b, (short)0, c, false, false);
}
static __device__ __forceinline__ v16h cat8(v8h lo, v8h hi) {
  return __builtin_shufflevector(lo, hi, 0,1,2,3,4,5,6,7,8,9,10,11,12,13,14,15);
}
static __device__ __forceinline__ v8h ldsv8(const _Float16* p) {
  return *(const v8h*)p;   // 16B-aligned by construction -> ds_load_b128
}

#if USE_TDM
// 1-D TDM load: nelem f16 elements, global -> LDS.  D# per CDNA5 ISA §8:
//  g0: [1:0]=count=1 | lds_addr[63:32] | global_addr[120:64] | type[127:126]=2
//  g1: data_size[17:16]=1(2B); tensor_dim0[79:48]; tile_dim0[127:112];
//      tensor_dim0_stride[207:160]
// This toolchain's builtin is the 6-arg form:
//   (uint32x4 g0, int32x8 g1, int32x4 g2, int32x4 g3, int32x8 extra, i32 cpol)
static __device__ __forceinline__ void tdm_load_1d(const void* gsrc,
                                                   void* lds_dst,
                                                   unsigned nelem) {
  const unsigned long long ga = (unsigned long long)gsrc;
  // addrspace(3) offset: LDS aperture flat addresses map to LDS by their
  // low 32 bits (ISA §10.2 aperture table).
  const unsigned lds_off = (unsigned)(unsigned long long)lds_dst;
  v4u g0;
  g0[0] = 1u;                                    // count=1, user mode
  g0[1] = lds_off;                               // lds_addr
  g0[2] = (unsigned)ga;                          // global_addr[31:0]
  g0[3] = (unsigned)(ga >> 32) | (2u << 30);     // global_addr[56:32] | type=2
  v8i g1;
  g1[0] = 0x00010000;                            // data_size=1 -> 2 bytes
  g1[1] = (int)((nelem & 0xFFFFu) << 16);        // tensor_dim0[15:0]
  g1[2] = (int)((nelem >> 16) & 0xFFFFu) | (1 << 16);  // tensor_dim0[31:16]; tensor_dim1=1
  g1[3] = (int)((nelem & 0xFFFFu) << 16);        // tile_dim0 (16-bit)
  g1[4] = 0;                                     // tile_dim1=0, tile_dim2=0
  g1[5] = (int)nelem;                            // tensor_dim0_stride[31:0]
  g1[6] = 0;
  g1[7] = 0;
  const v4i gz4 = {0, 0, 0, 0};
  const v8i gz8 = {0, 0, 0, 0, 0, 0, 0, 0};
  __builtin_amdgcn_tensor_load_to_lds(g0, g1, gz4, gz4, gz8, 0);
}
#endif

// ---------------------------------------------------------------------------
// Weight swizzle (once, in prep): W[K][128] -> fragment-major
// [(kstep*8+ntile)][lane][e]: each lane's 16-elem B fragment = 32 contig bytes.
//   e = (K%32>=16 ? 8:0) + (K%16) - 8h,  h = ((K%16)>=8),  lane = 16h + col%16
// ---------------------------------------------------------------------------
static __device__ __forceinline__ int frag_index(int K, int col) {
  const int n = col >> 4, cN = col & 15;
  const int kstep = K >> 5, kr = K & 31;
  const int hi16 = (kr >= 16) ? 8 : 0;
  const int kr16 = kr & 15;
  const int h = (kr16 >= 8) ? 1 : 0;
  const int e = hi16 + kr16 - 8 * h;
  return ((kstep * 8 + n) * 32 + (h * 16 + cN)) * 16 + e;
}

// ---------------------------------------------------------------------------
// Prep kernels
// ---------------------------------------------------------------------------
__global__ void prep_x_kernel(const long long* __restrict__ xcat,
                              const float* __restrict__ xfeat,
                              const float* __restrict__ emb,
                              _Float16* __restrict__ xh) {
  const int i = blockIdx.x * blockDim.x + threadIdx.x;  // 0..1023 = B*N
  if (i >= 4 * 256) return;
  const int cat = (int)xcat[i];
  const float* er = emb + (long)cat * 63;
#pragma unroll 7
  for (int d = 0; d < 63; ++d) xh[i * 64 + d] = (_Float16)er[d];
  xh[i * 64 + 63] = (_Float16)xfeat[i];
}

__global__ void prep_w_kernel(const float* __restrict__ W1,
                              const float* __restrict__ W2,
                              const float* __restrict__ W3,
                              _Float16* __restrict__ w1f,
                              _Float16* __restrict__ w2f,
                              _Float16* __restrict__ w3f,
                              float* __restrict__ outAcc) {
  const int idx = blockIdx.x * blockDim.x + threadIdx.x;
  if (idx < 8192) {
    w1f[frag_index(idx >> 7, idx & 127)] = (_Float16)W1[idx];
  } else if (idx < 24576) {
    const int j = idx - 8192;
    w2f[frag_index(j >> 7, j & 127)] = (_Float16)W2[j];
  } else if (idx < 40960) {
    const int j = idx - 24576;
    w3f[frag_index(j >> 7, j & 127)] = (_Float16)W3[j];
  } else if (idx < 41472) {
    outAcc[idx - 40960] = 0.f;            // 512 = B*128
  }
}

// ---------------------------------------------------------------------------
// Main fused kernel: 1024 WGs = (b, iTile, jTile), 256 thr = 8 waves.
// Each wave owns 32 pair-rows: 2 Mtiles x 8 Ntiles of f32 accumulators.
// ---------------------------------------------------------------------------

// Preload the whole B panel into VGPRs, then one hazard-free WMMA burst:
// no instruction between the WMMAs writes any WMMA source register, so no
// 4-coexec WAR nops; LDS latency amortized over the whole panel load.
template <int KSTEPS>
static __device__ __forceinline__ void mma_block(
    const v16h A0[KSTEPS], const v16h A1[KSTEPS],
    const _Float16* bp, v8f acc[2][8]) {
  constexpr int NF = KSTEPS * 8;
  v16h Bf[NF];
#pragma unroll
  for (int fi = 0; fi < NF; ++fi)
    Bf[fi] = cat8(ldsv8(bp + fi * 512), ldsv8(bp + fi * 512 + 8));
#pragma unroll
  for (int t = 0; t < 2; ++t)
#pragma unroll
    for (int n = 0; n < 8; ++n)
      acc[t][n] = (v8f){0.f, 0.f, 0.f, 0.f, 0.f, 0.f, 0.f, 0.f};
#pragma unroll
  for (int fi = 0; fi < NF; ++fi) {
    const int k = fi >> 3, n = fi & 7;
    acc[0][n] = wmma_f16(A0[k], Bf[fi], acc[0][n]);
    acc[1][n] = wmma_f16(A1[k], Bf[fi], acc[1][n]);
  }
}

// Full GEMM layer: preload A fragments (two contiguous 16B runs per lane,
// per the ISA 16-bit A layout), then the B-panel WMMA burst.
template <int KSTEPS>
static __device__ __forceinline__ void mma_layer(
    const _Float16* sA, const _Float16* sWf, int m0, int lane, v8f acc[2][8]) {
  const int kk = (lane & 16) ? 8 : 0;
  const _Float16* a0p = sA + (m0 + (lane & 15)) * LDH + kk;
  const _Float16* a1p = a0p + 16 * LDH;
  v16h A0[KSTEPS], A1[KSTEPS];
#pragma unroll
  for (int k = 0; k < KSTEPS; ++k) {
    A0[k] = cat8(ldsv8(a0p + k * 32), ldsv8(a0p + k * 32 + 16));
    A1[k] = cat8(ldsv8(a1p + k * 32), ldsv8(a1p + k * 32 + 16));
  }
  mma_block<KSTEPS>(A0, A1, sWf + lane * 16, acc);
}

// Bias + relu + f16 store of the wave's 32x128 C tile into row-major LDS.
static __device__ __forceinline__ void epilogue_store(
    _Float16* sH, const float* sBias, int m0, int lane, v8f acc[2][8]) {
  float bv[8];
#pragma unroll
  for (int n = 0; n < 8; ++n) bv[n] = sBias[n * 16 + (lane & 15)];
#pragma unroll
  for (int t = 0; t < 2; ++t) {
    _Float16* op = sH + (m0 + t * 16 + ((lane & 16) >> 1)) * LDH + (lane & 15);
#pragma unroll
    for (int n = 0; n < 8; ++n)
#pragma unroll
      for (int r = 0; r < 8; ++r)
        op[r * LDH + n * 16] = (_Float16)fmaxf(acc[t][n][r] + bv[n], 0.f);
  }
}

__global__ __launch_bounds__(256) void pair_mlp_kernel(
    const _Float16* __restrict__ xh,    // [B*N][64]
    const _Float16* __restrict__ w1f,   // frag-major [2*8][32][16]
    const _Float16* __restrict__ w2f,   // frag-major [4*8][32][16]
    const _Float16* __restrict__ w3f,
    const float* __restrict__ b1, const float* __restrict__ b2,
    const float* __restrict__ b3,
    float* __restrict__ outAcc)         // [B][128]
{
  __shared__ __attribute__((aligned(16))) _Float16 sW1[64 * 128];
  __shared__ __attribute__((aligned(16))) _Float16 sW2[128 * 128];
  __shared__ __attribute__((aligned(16))) _Float16 sW3[128 * 128];
  __shared__ __attribute__((aligned(16))) _Float16 sXi[16 * 64];
  __shared__ __attribute__((aligned(16))) _Float16 sXj[16 * 64];
  __shared__ __attribute__((aligned(16))) _Float16 sH[256 * LDH];
  __shared__ float sB1[128], sB2[128], sB3[128];
  __shared__ float sRed[128];

  const int tid  = threadIdx.x;
  const int lane = tid & 31;
  const int wave = tid >> 5;
  const int bx = blockIdx.x;
  const int b  = bx >> 8;
  const int iT = (bx >> 4) & 15;
  const int jT = bx & 15;

  // --- staging: weights via Tensor Data Mover (fallback: uint4 copies) ---
#if USE_TDM
  if (wave == 0) {
    tdm_load_1d(w1f, sW1, 64 * 128);
    tdm_load_1d(w2f, sW2, 128 * 128);
    tdm_load_1d(w3f, sW3, 128 * 128);
  }
#else
  {
    const uint4* g1 = (const uint4*)w1f;  uint4* s1 = (uint4*)sW1;
    const uint4* g2 = (const uint4*)w2f;  uint4* s2 = (uint4*)sW2;
    const uint4* g3 = (const uint4*)w3f;  uint4* s3 = (uint4*)sW3;
    for (int k = tid; k < 1024; k += 256) s1[k] = g1[k];
    for (int k = tid; k < 2048; k += 256) s2[k] = g2[k];
    for (int k = tid; k < 2048; k += 256) s3[k] = g3[k];
  }
#endif
  {
    const uint4* gx = (const uint4*)xh;   // 8 x 16B per 64-elem row
    uint4* sxi = (uint4*)sXi;  uint4* sxj = (uint4*)sXj;
    if (tid < 128) {
      const int r = tid >> 3, c = tid & 7;
      sxi[tid] = gx[(b * 256 + iT * 16 + r) * 8 + c];
      sxj[tid] = gx[(b * 256 + jT * 16 + r) * 8 + c];
      sB1[tid] = b1[tid]; sB2[tid] = b2[tid]; sB3[tid] = b3[tid];
      sRed[tid] = 0.f;
    }
  }
#if USE_TDM
  if (wave == 0) __builtin_amdgcn_s_wait_tensorcnt(0);  // wave 0 owns TENSORcnt
#endif
  __syncthreads();

  v8f acc[2][8];
  const int m0 = wave * 32;   // this wave's pair-row base

  // ---- layer 1: p(64)->128.  For a 16-row Mtile of pairs m = 16*iL + jL,
  // iL is constant and jL = lane&15 -> A = (xi_row broadcast) * xj_row,
  // built from contiguous 16B runs with v_pk_mul_f16. ----
  {
    const int kk = (lane & 16) ? 8 : 0;
    const _Float16* xj  = sXj + (lane & 15) * 64 + kk;
    const _Float16* xi0 = sXi + (wave * 2) * 64 + kk;   // iL for t=0
    const _Float16* xi1 = xi0 + 64;                     // iL for t=1
    v16h A0[2], A1[2];
#pragma unroll
    for (int k = 0; k < 2; ++k) {
      v8h jlo = ldsv8(xj + k * 32), jhi = ldsv8(xj + k * 32 + 16);
      A0[k] = cat8(ldsv8(xi0 + k * 32) * jlo, ldsv8(xi0 + k * 32 + 16) * jhi);
      A1[k] = cat8(ldsv8(xi1 + k * 32) * jlo, ldsv8(xi1 + k * 32 + 16) * jhi);
    }
    mma_block<2>(A0, A1, sW1 + lane * 16, acc);
  }
  epilogue_store(sH, sB1, m0, lane, acc);
  __syncthreads();

  // ---- layer 2: 128 -> 128 ----
  mma_layer<4>(sH, sW2, m0, lane, acc);
  epilogue_store(sH, sB2, m0, lane, acc);
  __syncthreads();

  // ---- layer 3: 128 -> 128, relu, pooled sum ----
  mma_layer<4>(sH, sW3, m0, lane, acc);
#pragma unroll
  for (int n = 0; n < 8; ++n) {
    const int col = n * 16 + (lane & 15);
    const float bv = sB3[col];
    float psum = 0.f;
#pragma unroll
    for (int t = 0; t < 2; ++t)
#pragma unroll
      for (int r = 0; r < 8; ++r)
        psum += fmaxf(acc[t][n][r] + bv, 0.f);
    atomicAdd(&sRed[col], psum);          // ds_add_f32
  }
  __syncthreads();
  if (tid < 128)
    atomicAdd(&outAcc[b * 128 + tid], sRed[tid] * (1.0f / 65536.0f));
}

// ---------------------------------------------------------------------------
// Decoder: [4,128] -> relu(D1) -> relu(D2) -> D3 -> [4,1].  Negligible work.
// ---------------------------------------------------------------------------
__global__ void decoder_kernel(const float* __restrict__ outAcc,
                               const float* __restrict__ Dw1, const float* __restrict__ c1,
                               const float* __restrict__ Dw2, const float* __restrict__ c2,
                               const float* __restrict__ Dw3, const float* __restrict__ c3,
                               float* __restrict__ out) {
  __shared__ float g[128], h1[128], h2[128], red[128];
  const int t = threadIdx.x;
  for (int b = 0; b < 4; ++b) {
    g[t] = fmaxf(outAcc[b * 128 + t], 0.f);   // relu after mean
    __syncthreads();
    float a = 0.f;
    for (int k = 0; k < 128; ++k) a += g[k] * Dw1[k * 128 + t];
    h1[t] = fmaxf(a + c1[t], 0.f);
    __syncthreads();
    a = 0.f;
    for (int k = 0; k < 128; ++k) a += h1[k] * Dw2[k * 128 + t];
    h2[t] = fmaxf(a + c2[t], 0.f);
    __syncthreads();
    red[t] = h2[t] * Dw3[t];
    __syncthreads();
    for (int s = 64; s > 0; s >>= 1) {
      if (t < s) red[t] += red[t + s];
      __syncthreads();
    }
    if (t == 0) out[b] = red[0] + c3[0];
    __syncthreads();
  }
}

// ---------------------------------------------------------------------------
extern "C" void kernel_launch(void* const* d_in, const int* in_sizes, int n_in,
                              void* d_out, int out_size, void* d_ws, size_t ws_size,
                              hipStream_t stream) {
  const long long* xcat  = (const long long*)d_in[0];
  const float* xfeat = (const float*)d_in[1];
  const float* emb   = (const float*)d_in[2];
  const float* W1 = (const float*)d_in[3];  const float* b1 = (const float*)d_in[4];
  const float* W2 = (const float*)d_in[5];  const float* b2 = (const float*)d_in[6];
  const float* W3 = (const float*)d_in[7];  const float* b3 = (const float*)d_in[8];
  const float* D1 = (const float*)d_in[9];  const float* c1 = (const float*)d_in[10];
  const float* D2 = (const float*)d_in[11]; const float* c2 = (const float*)d_in[12];
  const float* D3 = (const float*)d_in[13]; const float* c3 = (const float*)d_in[14];
  float* out = (float*)d_out;

  // workspace carve
  char* ws = (char*)d_ws;
  _Float16* xh  = (_Float16*)ws;                              // 131072 B
  _Float16* w1f = (_Float16*)(ws + 131072);                   //  16384 B
  _Float16* w2f = (_Float16*)(ws + 131072 + 16384);           //  32768 B
  _Float16* w3f = (_Float16*)(ws + 131072 + 16384 + 32768);   //  32768 B
  float* outAcc = (float*)(ws + 131072 + 16384 + 32768 + 32768);  // 2048 B

  prep_x_kernel<<<4, 256, 0, stream>>>(xcat, xfeat, emb, xh);
  prep_w_kernel<<<(41472 + 255) / 256, 256, 0, stream>>>(W1, W2, W3, w1f, w2f, w3f, outAcc);
  pair_mlp_kernel<<<4 * 16 * 16, 256, 0, stream>>>(xh, w1f, w2f, w3f, b1, b2, b3, outAcc);
  decoder_kernel<<<1, 128, 0, stream>>>(outAcc, D1, c1, D2, c2, D3, c3, out);
}
